// DeformableNet_893353197643
// MI455X (gfx1250) — matrix operand order repaired and checked by reading, not compiled, not run on hardware
//
#include <hip/hip_runtime.h>

typedef __attribute__((ext_vector_type(16))) _Float16 v16h;
typedef __attribute__((ext_vector_type(2)))  _Float16 v2h;
typedef __attribute__((ext_vector_type(8)))  float    v8f;
typedef __attribute__((ext_vector_type(2)))  float    v2f;

#define Bn   8
#define Cn   64
#define Hn   128
#define Wn   128
#define HW   (Hn * Wn)
#define K2n  9
#define KDIM (Cn * K2n)   // 576, ordered k = k2*64 + c
#define KB   (KDIM / 32)  // 18 WMMA k-steps
#define TILE_W 16
#define NTILES (Wn / TILE_W)
#define NITEMS (TILE_W * K2n * 2)  // 288 work items: (pixel,tap) x channel-half

// Gather one 16-pixel tile: bilinear sample 9 taps x 64 channels -> f16 B-matrix in LDS.
// buf[n*KDIM + k2*64 + c]; interior samples use paired-corner b64 loads.
__device__ __forceinline__
void gather_tile(_Float16* __restrict__ buf, int tw, int tid, int b, int h,
                 const float* __restrict__ abase, const float* __restrict__ off)
{
    const int w0 = tw * TILE_W;
    for (int it = tid; it < NITEMS; it += 128) {
        int pair  = it % (TILE_W * K2n);
        int chalf = it / (TILE_W * K2n);      // 0 or 1 -> channels [0,32) or [32,64)
        int np = pair / K2n;
        int k2 = pair - np * K2n;
        int w  = w0 + np;
        int ky = k2 / 3, kx = k2 - ky * 3;

        float oy = off[(((b * 18 + 2 * k2    ) * Hn + h) * Wn) + w];
        float ox = off[(((b * 18 + 2 * k2 + 1) * Hn + h) * Wn) + w];
        float py = (float)(h - 1 + ky) + oy;
        float px = (float)(w - 1 + kx) + ox;
        float fy = floorf(py), fx = floorf(px);
        float wy = py - fy,    wx = px - fx;
        int y0 = (int)fy, x0 = (int)fx;
        int y1 = y0 + 1,  x1 = x0 + 1;

        float w00 = (1.f - wy) * (1.f - wx);
        float w01 = (1.f - wy) * wx;
        float w10 = wy * (1.f - wx);
        float w11 = wy * wx;

        const int c0 = chalf * 32;
        _Float16* srow = &buf[np * KDIM + k2 * 64 + c0];

        if ((y0 >= 0) & (x0 >= 0) & (y1 < Hn) & (x1 < Wn)) {
            // interior: corners (x0,x0+1) contiguous -> one b64 per bilinear row
            const float* p0 = abase + c0 * HW + y0 * Wn + x0;  // row y0
            const float* p1 = p0 + Wn;                         // row y1
#pragma unroll 4
            for (int c2 = 0; c2 < 16; ++c2) {  // 2 channels per iteration
                v2f a0 = *reinterpret_cast<const v2f*>(p0);
                v2f a1 = *reinterpret_cast<const v2f*>(p1);
                v2f b0 = *reinterpret_cast<const v2f*>(p0 + HW);
                v2f b1 = *reinterpret_cast<const v2f*>(p1 + HW);
                float v0 = w00 * a0.x + w01 * a0.y + w10 * a1.x + w11 * a1.y;
                float v1 = w00 * b0.x + w01 * b0.y + w10 * b1.x + w11 * b1.y;
                v2h pk = { (_Float16)v0, (_Float16)v1 };
                *reinterpret_cast<v2h*>(&srow[2 * c2]) = pk;
                p0 += 2 * HW; p1 += 2 * HW;
            }
        } else {
            // border: zero-padded bilinear via zeroed corner weights + clamped loads
            bool vy0 = (y0 >= 0) & (y0 < Hn), vy1 = (y1 >= 0) & (y1 < Hn);
            bool vx0 = (x0 >= 0) & (x0 < Wn), vx1 = (x1 >= 0) & (x1 < Wn);
            float z00 = (vy0 && vx0) ? w00 : 0.f;
            float z01 = (vy0 && vx1) ? w01 : 0.f;
            float z10 = (vy1 && vx0) ? w10 : 0.f;
            float z11 = (vy1 && vx1) ? w11 : 0.f;
            int yc0 = min(max(y0, 0), Hn - 1), yc1 = min(max(y1, 0), Hn - 1);
            int xc0 = min(max(x0, 0), Wn - 1), xc1 = min(max(x1, 0), Wn - 1);
            int i00 = yc0 * Wn + xc0, i01 = yc0 * Wn + xc1;
            int i10 = yc1 * Wn + xc0, i11 = yc1 * Wn + xc1;
            const float* ac = abase + c0 * HW;
#pragma unroll 4
            for (int c2 = 0; c2 < 16; ++c2) {
                float v0 = z00 * ac[i00] + z01 * ac[i01] + z10 * ac[i10] + z11 * ac[i11];
                ac += HW;
                float v1 = z00 * ac[i00] + z01 * ac[i01] + z10 * ac[i10] + z11 * ac[i11];
                ac += HW;
                v2h pk = { (_Float16)v0, (_Float16)v1 };
                *reinterpret_cast<v2h*>(&srow[2 * c2]) = pk;
            }
        }
    }
}

__global__ __launch_bounds__(128)
void deform_conv_wmma(const float* __restrict__ a,
                      const float* __restrict__ off,
                      const float* __restrict__ weight,
                      const float* __restrict__ bias,
                      float* __restrict__ out)
{
    // A fragments for 4 o-tiles: [t][kb][lane][halfword] -> 36864 f16 = 72KB
    __shared__ __align__(32) _Float16 awgt[4 * KB * 32 * 16];
    // double-buffered sampled B matrix: 2 x 16x576 f16 = 2 x 18KB
    __shared__ __align__(32) _Float16 smat[2][TILE_W * KDIM];

    const int tid  = threadIdx.x;
    const int lane = tid & 31;
    const int wv   = tid >> 5;
    const int half = lane >> 4;
    const int z    = blockIdx.x;
    const int b    = z >> 7;
    const int h    = z & 127;

    // ---- reorder weights into f16 A-fragment layout (K ordered k2*64 + c) ----
    // lane l holds M = l&15; halfword i holds K = kb*32 + (l>>4)*16 + i
    for (int e = tid; e < 4 * KB * 32 * 16; e += 128) {
        int i  = e & 15;
        int l  = (e >> 4) & 31;
        int kb = (e >> 9) % KB;
        int t  = e / (KB * 32 * 16);
        int k  = kb * 32 + ((l >> 4) << 4) + i;
        int k2 = k >> 6;       // tap
        int c  = k & 63;       // input channel
        int o  = t * 16 + (l & 15);
        awgt[e] = (_Float16)weight[o * KDIM + c * K2n + k2];
    }

    // bias matching D layout (VGPR r -> M = r + 8*half)
    float bv[8];
#pragma unroll
    for (int r = 0; r < 8; ++r) bv[r] = bias[wv * 16 + half * 8 + r];

    const float* abase = a + b * (Cn * HW);
    const int n = lane & 15;

    __syncthreads();

    gather_tile(smat[0], 0, tid, b, h, abase, off);
    __syncthreads();

    for (int tw = 0; tw < NTILES; ++tw) {
        const int cur = tw & 1;

        // issue next tile's gather first: global-load latency overlaps WMMA below
        if (tw + 1 < NTILES)
            gather_tile(smat[cur ^ 1], tw + 1, tid, b, h, abase, off);

        // ---- GEMM: D(16x16) += A(16x32) * B(32x16), 18 k-steps ----
        v8f acc = {};
#pragma unroll
        for (int kb = 0; kb < KB; ++kb) {
            v16h af = *reinterpret_cast<const v16h*>(&awgt[((wv * KB + kb) * 32 + lane) * 16]);
            v16h bf = *reinterpret_cast<const v16h*>(&smat[cur][n * KDIM + kb * 32 + half * 16]);
            acc = __builtin_amdgcn_wmma_f32_16x16x32_f16(
                false, af, false, bf, (short)0, acc, false, false);
        }

        // ---- epilogue: bias + store (VGPR r holds M = r + 8*half, N = n) ----
        const int w0 = tw * TILE_W;
#pragma unroll
        for (int r = 0; r < 8; ++r) {
            int o = wv * 16 + half * 8 + r;
            out[(((b * Cn + o) * Hn + h) * Wn) + w0 + n] = acc[r] + bv[r];
        }
        // one barrier per tile: orders WMMA(t) reads before buffer reuse at t+2
        __syncthreads();
    }
}

extern "C" void kernel_launch(void* const* d_in, const int* in_sizes, int n_in,
                              void* d_out, int out_size, void* d_ws, size_t ws_size,
                              hipStream_t stream) {
    const float* a      = (const float*)d_in[0];
    const float* off    = (const float*)d_in[1];
    const float* weight = (const float*)d_in[2];
    const float* bias   = (const float*)d_in[3];
    float* out = (float*)d_out;
    (void)in_sizes; (void)n_in; (void)out_size; (void)d_ws; (void)ws_size;

    dim3 grid(Bn * Hn);   // 1024 blocks: one (b, h) row each
    dim3 block(128);      // 4 waves: one 16-channel o-tile per wave
    hipLaunchKernelGGL(deform_conv_wmma, grid, block, 0, stream,
                       a, off, weight, bias, out);
}
